// TemporalDecayAttention_18940805775707
// MI455X (gfx1250) — matrix operand order repaired and checked by reading, not compiled, not run on hardware
//
#include <hip/hip_runtime.h>
#include <hip/hip_bf16.h>

#define E_DIM 512
#define SEQ   2048
#define BATCH 4
#define NHEAD 8
#define HDIM  64
#define BHN   (BATCH * NHEAD)
#define MROWS (BATCH * SEQ)
#define NSTEP (E_DIM / 32)

typedef __attribute__((ext_vector_type(16))) __bf16 v16bf;
typedef __attribute__((ext_vector_type(8)))  float  v8f;

union FragBF { v16bf v; uint4 q[2]; };
union U4S    { uint4 u; unsigned short s[8]; };

__device__ __forceinline__ unsigned short f2bf(float f) {
  unsigned int u = __float_as_uint(f);
  u += 0x7FFFu + ((u >> 16) & 1u);   // round to nearest even
  return (unsigned short)(u >> 16);
}

// ---------------------------------------------------------------------------
// Tensor Data Mover 2D tile load (guarded: falls back to cooperative loads)
// ---------------------------------------------------------------------------
#if defined(__gfx1250__) && __has_builtin(__builtin_amdgcn_tensor_load_to_lds)
#define TDM_OK 1
#endif

#ifdef TDM_OK
typedef unsigned int u32x4 __attribute__((ext_vector_type(4)));
typedef int          i32x4 __attribute__((ext_vector_type(4)));
typedef int          i32x8 __attribute__((ext_vector_type(8)));

// D# built per CDNA5 ISA ch.8: group0 = {count/type/lds/global addr},
// group1 = {data_size, tensor dims, tile dims, dim0 stride}. 2D tensor ->
// groups 2/3 zero (tile_dim2 = 0 marks them unused).
__device__ __forceinline__ void tdm_load_2d(unsigned lds_off,
                                            const unsigned short* gptr,
                                            unsigned tensor_d0, unsigned tensor_d1,
                                            unsigned stride0_elems,
                                            unsigned tile_d0, unsigned tile_d1) {
  unsigned long long ga = (unsigned long long)gptr;
  u32x4 g0 = { 1u,                                                   // count=1
               lds_off,                                              // lds_addr (bytes)
               (unsigned)ga,                                         // global_addr[31:0]
               ((unsigned)((ga >> 32) & 0x01FFFFFFull)) | 0x80000000u };  // addr[56:32] | type=2
  i32x8 g1;
  g1[0] = (int)(1u << 16);                                           // data_size=1 (2 bytes)
  g1[1] = (int)((tensor_d0 & 0xFFFFu) << 16);                        // tensor_dim0 lo16
  g1[2] = (int)(((tensor_d0 >> 16) & 0xFFFFu) | ((tensor_d1 & 0xFFFFu) << 16));
  g1[3] = (int)(((tensor_d1 >> 16) & 0xFFFFu) | ((tile_d0 & 0xFFFFu) << 16));
  g1[4] = (int)(tile_d1 & 0xFFFFu);                                  // tile_dim1 (tile_dim2=0)
  g1[5] = (int)stride0_elems;                                        // tensor_dim0_stride lo32
  g1[6] = 0;
  g1[7] = 0;
  i32x4 z4 = {0, 0, 0, 0};
#if __clang_major__ >= 23
  i32x8 z8 = {0, 0, 0, 0, 0, 0, 0, 0};
  __builtin_amdgcn_tensor_load_to_lds(g0, g1, z4, z4, z8, 0);
#else
  __builtin_amdgcn_tensor_load_to_lds(g0, g1, z4, z4, 0);
#endif
}
#endif  // TDM_OK

// ---------------------------------------------------------------------------
// fp32 -> raw bf16 elementwise conversion
// ---------------------------------------------------------------------------
__global__ __launch_bounds__(256) void cvt_f32_bf16(unsigned short* __restrict__ dst,
                                                    const float* __restrict__ src,
                                                    int n) {
  int i = blockIdx.x * 256 + threadIdx.x;
  if (i < n) dst[i] = f2bf(src[i]);
}

// ---------------------------------------------------------------------------
// C = A(bf16 MxK) * W(bf16 NxK)^T + bias.  M=8192, N=K=512.
// Block: 256 threads (8 waves). Block tile 128(M) x 128(N); per wave 64x32
// (4x2 WMMA tiles -> 8 WMMA per K-step).  K-step 32, double-buffered LDS.
// Tile staging via TDM DMA (wave 0 issues, TENSORcnt-waited) when available.
// mode 0: store bf16 scattered into [B, H, S, 64] (q/k/v)
// mode 1: store fp32 into [M, 512] (final output)
// ---------------------------------------------------------------------------
__global__ __launch_bounds__(256) void gemm_bf16_512(
    const unsigned short* __restrict__ A,
    const unsigned short* __restrict__ W,
    const float* __restrict__ bias,
    unsigned short* __restrict__ outQKV,
    float* __restrict__ outF32,
    int mode)
{
  __shared__ __align__(16) unsigned short As[2][128][32];  // 2 x 8 KB
  __shared__ __align__(16) unsigned short Bs[2][128][32];  // 2 x 8 KB

  const int tid  = threadIdx.x;
  const int wave = tid >> 5;
  const int lane = tid & 31;
  const int half = lane >> 4;
  const int l16  = lane & 15;

  const int mBase = blockIdx.x * 128;
  const int nBase = blockIdx.y * 128;
  const int wm = (wave & 1) * 64;    // wave M offset in block tile
  const int wn = (wave >> 1) * 32;   // wave N offset in block tile

  v8f acc[4][2] = {};

#ifdef TDM_OK
  if (wave == 0) {
    tdm_load_2d((unsigned)(unsigned long long)(void*)&As[0][0][0],
                A + (size_t)mBase * E_DIM, E_DIM, MROWS, E_DIM, 32, 128);
    tdm_load_2d((unsigned)(unsigned long long)(void*)&Bs[0][0][0],
                W + (size_t)nBase * E_DIM, E_DIM, E_DIM, E_DIM, 32, 128);
  }
#endif

  for (int s = 0; s < NSTEP; ++s) {
    const int buf = s & 1;
    const int k0 = s * 32;
    __syncthreads();   // previous buffers' readers are done
#ifdef TDM_OK
    if (wave == 0) {
      if (s + 1 < NSTEP) {
        // prefetch next K-slab into the other buffer, then wait for current
        tdm_load_2d((unsigned)(unsigned long long)(void*)&As[buf ^ 1][0][0],
                    A + (size_t)mBase * E_DIM + k0 + 32, E_DIM, MROWS, E_DIM, 32, 128);
        tdm_load_2d((unsigned)(unsigned long long)(void*)&Bs[buf ^ 1][0][0],
                    W + (size_t)nBase * E_DIM + k0 + 32, E_DIM, E_DIM, E_DIM, 32, 128);
        __builtin_amdgcn_s_wait_tensorcnt(2);
      } else {
        __builtin_amdgcn_s_wait_tensorcnt(0);
      }
    }
#else
    // Cooperative staging: A/B tiles 128x32 bf16 each (512 uint4 chunks each)
    #pragma unroll
    for (int it = 0; it < 2; ++it) {
      int idx = tid + it * 256;
      int r = idx >> 2, c = idx & 3;
      *(uint4*)&As[buf][r][c * 8] =
          *(const uint4*)(A + (size_t)(mBase + r) * E_DIM + k0 + c * 8);
      *(uint4*)&Bs[buf][r][c * 8] =
          *(const uint4*)(W + (size_t)(nBase + r) * E_DIM + k0 + c * 8);
    }
#endif
    __syncthreads();   // tile visible to all waves

    // A layout (16x32 bf16): lane half 0 holds K {0..7, 16..23}, half 1 {8..15, 24..31}
    FragBF a[4], b[2];
    #pragma unroll
    for (int mi = 0; mi < 4; ++mi) {
      int row = wm + mi * 16 + l16;
      a[mi].q[0] = *(const uint4*)&As[buf][row][half * 8];
      a[mi].q[1] = *(const uint4*)&As[buf][row][16 + half * 8];
    }
    // B layout (32x16 bf16): lane holds column n = lane%16, half 0 K {0..7, 16..23}, half 1 {8..15, 24..31}
    #pragma unroll
    for (int ni = 0; ni < 2; ++ni) {
      int row = wn + ni * 16 + l16;
      b[ni].q[0] = *(const uint4*)&Bs[buf][row][half * 8];
      b[ni].q[1] = *(const uint4*)&Bs[buf][row][16 + half * 8];
    }
    #pragma unroll
    for (int mi = 0; mi < 4; ++mi)
      #pragma unroll
      for (int ni = 0; ni < 2; ++ni)
        acc[mi][ni] = __builtin_amdgcn_wmma_f32_16x16x32_bf16(
            false, a[mi].v, false, b[ni].v, (short)0, acc[mi][ni], false, false);
  }

  // Epilogue. C layout: VGPR j -> row j + 8*half, col = lane%16.
  #pragma unroll
  for (int mi = 0; mi < 4; ++mi)
    #pragma unroll
    for (int ni = 0; ni < 2; ++ni)
      #pragma unroll
      for (int j = 0; j < 8; ++j) {
        int gm = mBase + wm + mi * 16 + (half << 3) + j;
        int gn = nBase + wn + ni * 16 + l16;
        float v = acc[mi][ni][j] + bias[gn];
        if (mode == 0) {
          int b_ = gm >> 11, s_ = gm & 2047;   // gm = b*2048 + s
          int h_ = gn >> 6,  d_ = gn & 63;     // gn = h*64 + d
          outQKV[(((size_t)(b_ * NHEAD + h_) * SEQ + s_) << 6) + d_] = f2bf(v);
        } else {
          outF32[(size_t)gm * E_DIM + gn] = v;
        }
      }
}

// ---------------------------------------------------------------------------
// Flash-style attention with temporal decay.
// Block: 128 threads (4 waves), each wave owns 16 q rows; block covers 64 rows.
// Streams over 64-wide t-blocks with online softmax. All GEMMs via WMMA bf16.
// q/k/v: [BH, S, 64] bf16.  Output: [B, S, 512] bf16 (head-interleaved).
// ---------------------------------------------------------------------------
__global__ __launch_bounds__(128) void attn_decay_kernel(
    const unsigned short* __restrict__ qb,
    const unsigned short* __restrict__ kb,
    const unsigned short* __restrict__ vb,
    unsigned short* __restrict__ attnb)
{
  __shared__ __align__(16) unsigned short Vt[64][64];      // V transposed [d][t], 8 KB
  __shared__ __align__(16) unsigned short Ps[4][16][64];   // per-wave P staging, 8 KB

  const int tid  = threadIdx.x;
  const int wave = tid >> 5;
  const int lane = tid & 31;
  const int half = lane >> 4;
  const int l16  = lane & 15;

  const int bh = blockIdx.y;
  const int b  = bh >> 3, h = bh & 7;
  const int s0 = blockIdx.x * 64;
  const int sw = s0 + wave * 16;
  const size_t base = (size_t)bh * SEQ * HDIM;

  // Q fragments (M=16 rows, K=64 -> two K=32 chunks), loaded once.
  FragBF qf[2];
  {
    const unsigned short* qrow = qb + base + (size_t)(sw + l16) * HDIM;
    #pragma unroll
    for (int kc = 0; kc < 2; ++kc) {
      qf[kc].q[0] = *(const uint4*)(qrow + kc * 32 + half * 8);
      qf[kc].q[1] = *(const uint4*)(qrow + kc * 32 + 16 + half * 8);
    }
  }

  v8f o[4] = {};
  float mrow[8], lrow[8];
  #pragma unroll
  for (int j = 0; j < 8; ++j) { mrow[j] = -3.0e38f; lrow[j] = 0.f; }

  const float kLog2Decay = -0.07400058144377693f;  // log2(0.95)
  const float kScale = 0.125f;                     // 1/sqrt(64)

  for (int tb = 0; tb < SEQ / 64; ++tb) {
    const int t0 = tb * 64;
    __syncthreads();
    // Stage V tile transposed into LDS: Vt[d][t].
    #pragma unroll
    for (int it = 0; it < 4; ++it) {
      int idx = tid + it * 128;
      int t = idx >> 3, c = idx & 7;
      U4S u;
      u.u = *(const uint4*)(vb + base + (size_t)(t0 + t) * HDIM + c * 8);
      #pragma unroll
      for (int e = 0; e < 8; ++e) Vt[c * 8 + e][t] = u.s[e];
    }
    __syncthreads();

    // Scores: S = Q (16x64) * K^T (64 x 64t). K-frag columns = k rows (contiguous).
    float sv[4][8];
    float rmax[8];
    #pragma unroll
    for (int j = 0; j < 8; ++j) rmax[j] = -3.0e38f;

    #pragma unroll
    for (int nt = 0; nt < 4; ++nt) {
      const unsigned short* krow = kb + base + (size_t)(t0 + nt * 16 + l16) * HDIM;
      FragBF kf0, kf1;
      kf0.q[0] = *(const uint4*)(krow + half * 8);
      kf0.q[1] = *(const uint4*)(krow + 16 + half * 8);
      kf1.q[0] = *(const uint4*)(krow + 32 + half * 8);
      kf1.q[1] = *(const uint4*)(krow + 32 + 16 + half * 8);
      v8f z = {};
      z = __builtin_amdgcn_wmma_f32_16x16x32_bf16(false, qf[0].v, false, kf0.v, (short)0, z, false, false);
      z = __builtin_amdgcn_wmma_f32_16x16x32_bf16(false, qf[1].v, false, kf1.v, (short)0, z, false, false);
      int tcol = t0 + nt * 16 + l16;
      #pragma unroll
      for (int j = 0; j < 8; ++j) {
        int srow = sw + (half << 3) + j;
        float dd = fabsf((float)(srow - tcol));
        float val = z[j] * kScale * exp2f(kLog2Decay * dd);
        sv[nt][j] = val;
        rmax[j] = fmaxf(rmax[j], val);
      }
    }

    // Row max across the 16 lanes of each half (offsets 1..8 stay in-half).
    #pragma unroll
    for (int j = 0; j < 8; ++j) {
      #pragma unroll
      for (int off = 1; off < 16; off <<= 1)
        rmax[j] = fmaxf(rmax[j], __shfl_xor(rmax[j], off, 32));
    }

    float corr[8], rsum[8];
    #pragma unroll
    for (int j = 0; j < 8; ++j) {
      float nm = fmaxf(mrow[j], rmax[j]);
      corr[j] = __expf(mrow[j] - nm);
      mrow[j] = nm;
      rsum[j] = 0.f;
    }
    #pragma unroll
    for (int nt = 0; nt < 4; ++nt)
      #pragma unroll
      for (int j = 0; j < 8; ++j) {
        float p = __expf(sv[nt][j] - mrow[j]);
        sv[nt][j] = p;
        rsum[j] += p;
      }
    #pragma unroll
    for (int j = 0; j < 8; ++j) {
      #pragma unroll
      for (int off = 1; off < 16; off <<= 1)
        rsum[j] += __shfl_xor(rsum[j], off, 32);
      lrow[j] = lrow[j] * corr[j] + rsum[j];
    }
    #pragma unroll
    for (int nt = 0; nt < 4; ++nt)
      #pragma unroll
      for (int j = 0; j < 8; ++j)
        o[nt][j] *= corr[j];

    // Restage P (C layout -> A layout) through LDS as bf16.
    #pragma unroll
    for (int nt = 0; nt < 4; ++nt)
      #pragma unroll
      for (int j = 0; j < 8; ++j)
        Ps[wave][(half << 3) + j][nt * 16 + l16] = f2bf(sv[nt][j]);
    __syncthreads();

    FragBF pf[2];
    {
      const unsigned short* prow = &Ps[wave][l16][0];
      #pragma unroll
      for (int kc = 0; kc < 2; ++kc) {
        pf[kc].q[0] = *(const uint4*)(prow + kc * 32 + half * 8);
        pf[kc].q[1] = *(const uint4*)(prow + kc * 32 + 16 + half * 8);
      }
    }
    // O += P (16 x 64t) * V (64t x 64d). V B-frag columns = Vt rows (contiguous).
    #pragma unroll
    for (int nt = 0; nt < 4; ++nt) {
      const unsigned short* vrow = &Vt[nt * 16 + l16][0];
      FragBF vf0, vf1;
      vf0.q[0] = *(const uint4*)(vrow + half * 8);
      vf0.q[1] = *(const uint4*)(vrow + 16 + half * 8);
      vf1.q[0] = *(const uint4*)(vrow + 32 + half * 8);
      vf1.q[1] = *(const uint4*)(vrow + 32 + 16 + half * 8);
      o[nt] = __builtin_amdgcn_wmma_f32_16x16x32_bf16(false, pf[0].v, false, vf0.v, (short)0, o[nt], false, false);
      o[nt] = __builtin_amdgcn_wmma_f32_16x16x32_bf16(false, pf[1].v, false, vf1.v, (short)0, o[nt], false, false);
    }
  }

  // Normalize and store to [B, S, 512] with e = h*64 + d.
  float invl[8];
  #pragma unroll
  for (int j = 0; j < 8; ++j) invl[j] = 1.0f / lrow[j];
  #pragma unroll
  for (int nt = 0; nt < 4; ++nt)
    #pragma unroll
    for (int j = 0; j < 8; ++j) {
      int srow = sw + (half << 3) + j;
      int e = h * HDIM + nt * 16 + l16;
      attnb[((size_t)(b * SEQ + srow)) * E_DIM + e] = f2bf(o[nt][j] * invl[j]);
    }
}

// ---------------------------------------------------------------------------
extern "C" void kernel_launch(void* const* d_in, const int* in_sizes, int n_in,
                              void* d_out, int out_size, void* d_ws, size_t ws_size,
                              hipStream_t stream) {
  (void)in_sizes; (void)n_in; (void)out_size; (void)ws_size;

  const float* x  = (const float*)d_in[0];
  const float* Wq = (const float*)d_in[1];
  const float* bq = (const float*)d_in[2];
  const float* Wk = (const float*)d_in[3];
  const float* bk = (const float*)d_in[4];
  const float* Wv = (const float*)d_in[5];
  const float* bv = (const float*)d_in[6];
  const float* Wo = (const float*)d_in[7];
  const float* bo = (const float*)d_in[8];

  unsigned short* ws = (unsigned short*)d_ws;
  const size_t XN = (size_t)MROWS * E_DIM;   // 4,194,304 elems
  const size_t WN = (size_t)E_DIM * E_DIM;   //   262,144 elems
  unsigned short* xb  = ws;
  unsigned short* Wqb = xb + XN;
  unsigned short* Wkb = Wqb + WN;
  unsigned short* Wvb = Wkb + WN;
  unsigned short* Wob = Wvb + WN;
  unsigned short* qb  = Wob + WN;
  unsigned short* kbp = qb + XN;
  unsigned short* vbp = kbp + XN;
  unsigned short* ab  = vbp + XN;
  // total ws use: 5*XN + 4*WN elems = ~44 MB

  cvt_f32_bf16<<<(int)(XN / 256), 256, 0, stream>>>(xb, x, (int)XN);
  cvt_f32_bf16<<<(int)(WN / 256), 256, 0, stream>>>(Wqb, Wq, (int)WN);
  cvt_f32_bf16<<<(int)(WN / 256), 256, 0, stream>>>(Wkb, Wk, (int)WN);
  cvt_f32_bf16<<<(int)(WN / 256), 256, 0, stream>>>(Wvb, Wv, (int)WN);
  cvt_f32_bf16<<<(int)(WN / 256), 256, 0, stream>>>(Wob, Wo, (int)WN);

  dim3 gg(MROWS / 128, E_DIM / 128);
  gemm_bf16_512<<<gg, 256, 0, stream>>>(xb, Wqb, bq, qb,  nullptr, 0);
  gemm_bf16_512<<<gg, 256, 0, stream>>>(xb, Wkb, bk, kbp, nullptr, 0);
  gemm_bf16_512<<<gg, 256, 0, stream>>>(xb, Wvb, bv, vbp, nullptr, 0);

  attn_decay_kernel<<<dim3(SEQ / 64, BHN), 128, 0, stream>>>(qb, kbp, vbp, ab);

  gemm_bf16_512<<<gg, 256, 0, stream>>>(ab, Wob, bo, nullptr, (float*)d_out, 1);
}
